// LSTMModel_77824807404126
// MI455X (gfx1250) — compile-verified, matrix operand order checked by reading, not run on hardware
//
#include <hip/hip_runtime.h>
#include <hip/hip_bf16.h>
#include <stdint.h>

// ---------------------------------------------------------------------------
// 4-layer bidirectional LSTM, H=64, B=128, T=1024, fused WMMA recurrence.
// One workgroup = (direction, 16-batch-row group). 8 waves, each wave owns
// two 16-wide gate-column tiles; Whh / Wih live in VGPRs in WMMA B-layout
// for the entire T loop. Branch-free TRANS-unit activations, LDS-only
// lightweight split barriers (global stores drain off the critical path),
// and software-pipelined x-tile loads (fetched one timestep ahead).
// Activations ping-pong between two bf16 [B,T,128] buffers in d_ws (64 MB,
// resident in MI455X's 192 MB L2).
// ---------------------------------------------------------------------------

typedef __attribute__((ext_vector_type(16))) __bf16 v16bf;
typedef __attribute__((ext_vector_type(8)))  float  v8f;

#define B_   128
#define T_   1024
#define H_   64
#define G_   256   // 4*H gates
#define ROWS 16    // batch rows per workgroup

#define LOG2E 1.4426950408889634f

__device__ inline unsigned short f2bfu(float f) {
    unsigned u = __builtin_bit_cast(unsigned, f);
    return (unsigned short)((u + 0x7FFFu + ((u >> 16) & 1u)) >> 16);
}
__device__ inline __bf16 f2bf(float f) {
    unsigned short h = f2bfu(f);
    return __builtin_bit_cast(__bf16, h);
}
__device__ inline float bfu2f(unsigned short s) {
    unsigned u = ((unsigned)s) << 16;
    return __builtin_bit_cast(float, u);
}

// Branch-free activations on the TRANS units (co-execute with 16-bit WMMA).
__device__ inline float fast_sigmoid(float x) {
    return __builtin_amdgcn_rcpf(1.0f + __builtin_amdgcn_exp2f(-LOG2E * x));
}
#if __has_builtin(__builtin_amdgcn_tanhf)
__device__ inline float fast_tanh(float x) { return __builtin_amdgcn_tanhf(x); }
#else
__device__ inline float fast_tanh(float x) {
    // tanh(x) = 1 - 2/(1+e^{2x}); exp2 saturates cleanly to +-1 at the ends
    return 1.0f - 2.0f * __builtin_amdgcn_rcpf(1.0f + __builtin_amdgcn_exp2f(2.0f * LOG2E * x));
}
#endif

// Workgroup barrier ordering only LDS traffic: do NOT wait for global
// storecnt (out_act stores are never read inside this kernel; the implicit
// S_ENDPGM wait + device-scope L2 make them visible to the next launch).
__device__ inline void lds_barrier() {
    asm volatile("s_wait_dscnt 0x0\n\t"
                 "s_barrier_signal -1\n\t"
                 "s_barrier_wait -1" ::: "memory");
}

// WMMA bf16 A/B 16x16x32 per-lane layout:
//   lanes 0-15 : row = lane,     K = {kb+0..7, kb+16..23}, kb = kchunk*32
//   lanes 16-31: row = lane-16,  K = {kb+8..15, kb+24..31}
// Both 8-element K segments are 16B-contiguous -> two b128 loads per tile.

// B-tile from f32 weight matrix W[G][ld] (B = W^T, so B[k][n] = W[n][k]); runs once.
__device__ inline v16bf load_wtile(const float* W, int ld, int ntile, int kchunk, int lane) {
    int n  = ntile * 16 + (lane & 15);
    int kb = kchunk * 32 + ((lane & 16) ? 8 : 0);
    const float* row = W + (size_t)n * ld;
    v16bf r;
#pragma unroll
    for (int v = 0; v < 8; ++v) {
        int k = kb + (v < 4 ? 2 * v : 16 + 2 * (v - 4));
        r[2 * v]     = f2bf(row[k]);
        r[2 * v + 1] = f2bf(row[k + 1]);
    }
    return r;
}

union ATile {
    uint4 u[2];
    v16bf v;
};

// A-tile from LDS bf16 h state [ROWS][H_] -> 2x ds_load_b128
__device__ inline v16bf load_atile_lds(const unsigned short* hs, int kchunk, int lane) {
    int r0 = lane & 15;
    int kb = kchunk * 32 + ((lane & 16) ? 8 : 0);
    const unsigned short* row = hs + r0 * H_;
    ATile r;
    r.u[0] = *(const uint4*)(row + kb);
    r.u[1] = *(const uint4*)(row + kb + 16);
    return r.v;
}

// A-tile from a global bf16 activation row; branch-free invalid->zero via mask
__device__ inline v16bf load_atile_g(const unsigned short* rowp, int kchunk, int lane,
                                     unsigned mask) {
    int kb = kchunk * 32 + ((lane & 16) ? 8 : 0);
    ATile r;
    r.u[0] = *(const uint4*)(rowp + kb);
    r.u[1] = *(const uint4*)(rowp + kb + 16);
#pragma unroll
    for (int i = 0; i < 2; ++i) {
        r.u[i].x &= mask; r.u[i].y &= mask; r.u[i].z &= mask; r.u[i].w &= mask;
    }
    return r.v;
}

#define WMMA_BF16(A, Bm, C) \
    __builtin_amdgcn_wmma_f32_16x16x32_bf16(false, (A), false, (Bm), (short)0, (C), false, false)

template <int IN_D>
__global__ __launch_bounds__(256)
void lstm_layer_kernel(const unsigned short* __restrict__ in_act,  // bf16 [B][T][128] (IN_D==128)
                       const float* __restrict__ x0,               // f32  [B][T]      (IN_D==1)
                       const int* __restrict__ lengths,            // [B]
                       const float* __restrict__ Wih,              // [2][G][IN_D]
                       const float* __restrict__ Whh,              // [2][G][H]
                       const float* __restrict__ bias,              // [2][G]
                       unsigned short* __restrict__ out_act)       // bf16 [B][T][128]
{
    __shared__ __align__(16) float          gates[ROWS][G_];  // 16 KB
    __shared__ __align__(16) unsigned short hbf[ROWS][H_];    // 2 KB (bf16 h state)

    const int  tid  = threadIdx.x;
    const int  wave = tid >> 5;
    const int  lane = tid & 31;
    const int  dir  = blockIdx.x >> 3;  // 0 = fwd, 1 = bwd
    const int  bg   = blockIdx.x & 7;
    const int  b0   = bg * ROWS;
    const bool fwd  = (dir == 0);

    const float* Wih_d  = Wih  + (size_t)dir * G_ * IN_D;
    const float* Whh_d  = Whh  + (size_t)dir * G_ * H_;
    const float* bias_d = bias + dir * G_;

    // ---- resident weights in WMMA B-layout ----
    const int nt0 = wave * 2, nt1 = wave * 2 + 1;
    v16bf whB[2][2];
#pragma unroll
    for (int kc = 0; kc < 2; ++kc) {
        whB[0][kc] = load_wtile(Whh_d, H_, nt0, kc, lane);
        whB[1][kc] = load_wtile(Whh_d, H_, nt1, kc, lane);
    }
    v16bf wiB[2][4];
    if (IN_D == 128) {
#pragma unroll
        for (int kc = 0; kc < 4; ++kc) {
            wiB[0][kc] = load_wtile(Wih_d, IN_D, nt0, kc, lane);
            wiB[1][kc] = load_wtile(Wih_d, IN_D, nt1, kc, lane);
        }
    }

    // ---- per-thread elementwise ownership: row re = tid>>4, cols j0..j0+3 ----
    const int re = tid >> 4;            // 0..15 (constant across q)
    const int j0 = (tid & 15) * 4;      // 0,4,...,60
    float4 bi  = *(const float4*)(bias_d + 0 * H_ + j0);
    float4 bf  = *(const float4*)(bias_d + 1 * H_ + j0);
    float4 bg4 = *(const float4*)(bias_d + 2 * H_ + j0);
    float4 bo  = *(const float4*)(bias_d + 3 * H_ + j0);
    float4 wi0 = {}, wf0 = {}, wg0 = {}, wo0 = {};
    if (IN_D == 1) {
        wi0 = *(const float4*)(Wih_d + 0 * H_ + j0);
        wf0 = *(const float4*)(Wih_d + 1 * H_ + j0);
        wg0 = *(const float4*)(Wih_d + 2 * H_ + j0);
        wo0 = *(const float4*)(Wih_d + 3 * H_ + j0);
    }
    const int len_e = lengths[b0 + re];          // elementwise row length
    const int len_a = lengths[b0 + (lane & 15)]; // A-tile row length (phase 1)
    const int r0    = lane & 15;

    // h0 = 0
    for (int i = tid; i < ROWS * H_ / 4; i += 256) ((uint2*)&hbf[0][0])[i] = make_uint2(0u, 0u);
    float creg[4];
#pragma unroll
    for (int q = 0; q < 4; ++q) creg[q] = 0.0f;  // c0 = 0
    __syncthreads();

    // ---- software-pipelined per-step inputs (depend only on s, not on h) ----
    auto load_xtiles = [&](int s, v16bf (&ax)[4]) {
        int      t     = fwd ? s : (len_a - 1 - s);
        bool     valid = (fwd ? (s < T_) : (s < len_a));
        int      tc    = t < 0 ? 0 : (t > T_ - 1 ? T_ - 1 : t);
        unsigned mask  = valid ? 0xFFFFFFFFu : 0u;
        const unsigned short* rowp = in_act + (((size_t)(b0 + r0) * T_ + tc) << 7);
#pragma unroll
        for (int kc = 0; kc < 4; ++kc) ax[kc] = load_atile_g(rowp, kc, lane, mask);
    };
    auto load_xscalar = [&](int s) -> float {
        int  t     = fwd ? s : (len_e - 1 - s);
        bool valid = (fwd ? (s < T_) : (s < len_e));
        int  tc    = t < 0 ? 0 : (t > T_ - 1 ? T_ - 1 : t);
        float v    = x0[(size_t)(b0 + re) * T_ + tc];
        return valid ? v : 0.0f;
    };

    // one full LSTM timestep; consumes `cur`, prefetches s+1 into `nxt`
    auto step = [&](int s, v16bf (&cur)[4], v16bf (&nxt)[4], float xcur, float& xnxt) {
        // kick off next-step input loads first (independent of the recurrence)
        if (IN_D == 128) load_xtiles(s + 1, nxt);
        if (IN_D == 1)   xnxt = load_xscalar(s + 1);

        // ---------------- phase 1: gate GEMM -> LDS ----------------
        v16bf ah0 = load_atile_lds(&hbf[0][0], 0, lane);
        v16bf ah1 = load_atile_lds(&hbf[0][0], 1, lane);
        v8f acc0 = {}, acc1 = {};
        acc0 = WMMA_BF16(ah0, whB[0][0], acc0);
        acc0 = WMMA_BF16(ah1, whB[0][1], acc0);
        acc1 = WMMA_BF16(ah0, whB[1][0], acc1);
        acc1 = WMMA_BF16(ah1, whB[1][1], acc1);
        if (IN_D == 128) {
#pragma unroll
            for (int kc = 0; kc < 4; ++kc) {
                acc0 = WMMA_BF16(cur[kc], wiB[0][kc], acc0);
                acc1 = WMMA_BF16(cur[kc], wiB[1][kc], acc1);
            }
        }
        // scatter C tiles (f32 16x16 layout) to gates LDS
        {
            int rr = (lane & 16) ? 8 : 0;
            int cc = lane & 15;
#pragma unroll
            for (int i = 0; i < 8; ++i) {
                gates[i + rr][nt0 * 16 + cc] = acc0[i];
                gates[i + rr][nt1 * 16 + cc] = acc1[i];
            }
        }
        lds_barrier();

        // ---------------- phase 2: elementwise LSTM cell ----------------
        {
            int  t     = fwd ? s : (len_e - 1 - s);
            bool valid = fwd || (s < len_e);

            float4 gi = *(const float4*)(&gates[re][0 * H_ + j0]);
            float4 gf = *(const float4*)(&gates[re][1 * H_ + j0]);
            float4 gg = *(const float4*)(&gates[re][2 * H_ + j0]);
            float4 go = *(const float4*)(&gates[re][3 * H_ + j0]);
            gi.x += bi.x;  gi.y += bi.y;  gi.z += bi.z;  gi.w += bi.w;
            gf.x += bf.x;  gf.y += bf.y;  gf.z += bf.z;  gf.w += bf.w;
            gg.x += bg4.x; gg.y += bg4.y; gg.z += bg4.z; gg.w += bg4.w;
            go.x += bo.x;  go.y += bo.y;  go.z += bo.z;  go.w += bo.w;
            if (IN_D == 1) {
                float xv = xcur;
                gi.x += xv * wi0.x; gi.y += xv * wi0.y; gi.z += xv * wi0.z; gi.w += xv * wi0.w;
                gf.x += xv * wf0.x; gf.y += xv * wf0.y; gf.z += xv * wf0.z; gf.w += xv * wf0.w;
                gg.x += xv * wg0.x; gg.y += xv * wg0.y; gg.z += xv * wg0.z; gg.w += xv * wg0.w;
                go.x += xv * wo0.x; go.y += xv * wo0.y; go.z += xv * wo0.z; go.w += xv * wo0.w;
            }
            float gia[4] = {gi.x, gi.y, gi.z, gi.w};
            float gfa[4] = {gf.x, gf.y, gf.z, gf.w};
            float gga[4] = {gg.x, gg.y, gg.z, gg.w};
            float goa[4] = {go.x, go.y, go.z, go.w};
            unsigned short hb[4];
#pragma unroll
            for (int q = 0; q < 4; ++q) {
                float i_ = fast_sigmoid(gia[q]);
                float f_ = fast_sigmoid(gfa[q]);
                float g_ = fast_tanh(gga[q]);
                float o_ = fast_sigmoid(goa[q]);
                float c_ = f_ * creg[q] + i_ * g_;
                float h_ = o_ * fast_tanh(c_);
                creg[q] = c_;
                hb[q]   = f2bfu(h_);
            }
            uint2 packed = make_uint2((unsigned)hb[0] | ((unsigned)hb[1] << 16),
                                      (unsigned)hb[2] | ((unsigned)hb[3] << 16));
            *(uint2*)(&hbf[re][j0]) = packed;  // next step's A matrix
            if (valid) {
                *(uint2*)(out_act + (((size_t)(b0 + re) * T_ + t) << 7) + dir * H_ + j0) = packed;
            }
            if (!fwd && s >= len_e) {  // zero padded tail of the bwd half
                *(uint2*)(out_act + (((size_t)(b0 + re) * T_ + s) << 7) + H_ + j0) =
                    make_uint2(0u, 0u);
            }
        }
        lds_barrier();
    };

    v16bf axA[4], axB[4];
    float xvA = 0.0f, xvB = 0.0f;
    if (IN_D == 128) load_xtiles(0, axA);
    if (IN_D == 1)   xvA = load_xscalar(0);

    for (int s = 0; s < T_; s += 2) {
        step(s,     axA, axB, xvA, xvB);
        step(s + 1, axB, axA, xvB, xvA);
    }
}

__global__ __launch_bounds__(128)
void readout_kernel(const unsigned short* __restrict__ act,  // bf16 [B][T][128]
                    const int* __restrict__ lengths,
                    const float* __restrict__ Wh,             // [1][128]
                    const float* __restrict__ bh,             // [1]
                    float* __restrict__ out)                  // [B]
{
    int b = blockIdx.x * blockDim.x + threadIdx.x;
    if (b >= B_) return;
    int t = lengths[b] - 1;
    const unsigned short* row = act + (((size_t)b * T_ + t) << 7);
    float s = 0.0f;
#pragma unroll 4
    for (int j = 0; j < 128; ++j) s += bfu2f(row[j]) * Wh[j];
    out[b] = s + bh[0];
}

extern "C" void kernel_launch(void* const* d_in, const int* in_sizes, int n_in,
                              void* d_out, int out_size, void* d_ws, size_t ws_size,
                              hipStream_t stream) {
    const float* x       = (const float*)d_in[0];  // [128,1024,1]
    const int*   lengths = (const int*)d_in[1];    // [128]
    const float* W0      = (const float*)d_in[2];  // [2,256,1]
    const float* U0      = (const float*)d_in[3];  // [2,256,64]
    const float* b0      = (const float*)d_in[4];  // [2,256]
    const float* WL      = (const float*)d_in[5];  // [3,2,256,128]
    const float* UL      = (const float*)d_in[6];  // [3,2,256,64]
    const float* bL      = (const float*)d_in[7];  // [3,2,256]
    const float* Wh      = (const float*)d_in[8];  // [1,128]
    const float* bh      = (const float*)d_in[9];  // [1]
    float* out = (float*)d_out;

    unsigned short* actA = (unsigned short*)d_ws;                 // bf16 [B,T,128] = 32 MB
    unsigned short* actB = actA + (size_t)B_ * T_ * 128;          // +32 MB

    dim3 grid(16), blk(256);  // 2 dirs x 8 batch groups; 8 wave32s per WG

    // layer 0 (D=1)
    lstm_layer_kernel<1><<<grid, blk, 0, stream>>>(nullptr, x, lengths, W0, U0, b0, actA);
    // layers 1..3 (D=128), ping-pong activations
    lstm_layer_kernel<128><<<grid, blk, 0, stream>>>(
        actA, nullptr, lengths, WL + 0 * 2 * G_ * 128, UL + 0 * 2 * G_ * H_, bL + 0 * 2 * G_, actB);
    lstm_layer_kernel<128><<<grid, blk, 0, stream>>>(
        actB, nullptr, lengths, WL + 1 * 2 * G_ * 128, UL + 1 * 2 * G_ * H_, bL + 1 * 2 * G_, actA);
    lstm_layer_kernel<128><<<grid, blk, 0, stream>>>(
        actA, nullptr, lengths, WL + 2 * 2 * G_ * 128, UL + 2 * 2 * G_ * H_, bL + 2 * 2 * G_, actB);
    // final linear readout at t = len-1
    readout_kernel<<<1, 128, 0, stream>>>(actB, lengths, Wh, bh, out);
}
